// AttentionModule_9079560863766
// MI455X (gfx1250) — matrix operand order repaired and checked by reading, not compiled, not run on hardware
//
#include <hip/hip_runtime.h>
#include <cstddef>

// ---------------- types & helpers ----------------
typedef __attribute__((ext_vector_type(16))) __bf16 bf16x16;
typedef __attribute__((ext_vector_type(8)))  float  f32x8;
typedef __attribute__((ext_vector_type(4)))  unsigned int u32x4;

#define DEVINL __device__ __forceinline__

union FragU {
    bf16x16 v;
    u32x4   q[2];
    unsigned short h[16];
};

DEVINL unsigned short f2bf(float f) {
    unsigned int u = __float_as_uint(f);
    u += 0x7fffu + ((u >> 16) & 1u);
    return (unsigned short)(u >> 16);
}

// A fragment (16x32 bf16): lane L(0-15) holds row M=L, K = {0..7, 16..23}+k0;
// lane L+16 holds row M=L, K = {8..15, 24..31}+k0.  rowBase points at k=0 of the lane's row.
DEVINL bf16x16 loadA(const unsigned short* rowBase, int k0, int lane) {
    int d = (lane & 16) ? 8 : 0;
    FragU f;
    f.q[0] = *(const u32x4*)(rowBase + k0 + d);
    f.q[1] = *(const u32x4*)(rowBase + k0 + d + 16);
    return f.v;
}

// B fragment (32x16 bf16): lane L(0-15) holds col N=L, K = {0..7, 16..23}+k0;
// lane L+16 holds col N=L, K = {8..15, 24..31}+k0.  rowBase points at k=0 of the lane's N-row
// in a [N][K] (k-contiguous) buffer.
DEVINL bf16x16 loadB(const unsigned short* rowBase, int k0, int lane) {
    int d = (lane & 16) ? 8 : 0;
    FragU f;
    f.q[0] = *(const u32x4*)(rowBase + k0 + d);
    f.q[1] = *(const u32x4*)(rowBase + k0 + d + 16);
    return f.v;
}

// strided gathers (for the tiny channel-energy GEMM where K is the row dim)
DEVINL bf16x16 loadA_strided(const unsigned short* base, int stride, int k0, int lane) {
    int d = (lane & 16) ? 8 : 0;
    FragU f;
#pragma unroll
    for (int j = 0; j < 8; ++j) {
        f.h[j]     = base[(size_t)(k0 + d + j) * stride];
        f.h[8 + j] = base[(size_t)(k0 + 16 + d + j) * stride];
    }
    return f.v;
}
DEVINL bf16x16 loadB_strided(const unsigned short* base, int stride, int k0, int lane) {
    int d = (lane & 16) ? 8 : 0;
    FragU f;
#pragma unroll
    for (int j = 0; j < 8; ++j) {
        f.h[j]     = base[(size_t)(k0 + d + j) * stride];
        f.h[8 + j] = base[(size_t)(k0 + 16 + d + j) * stride];
    }
    return f.v;
}

DEVINL f32x8 wmma_bf16(bf16x16 a, bf16x16 b, f32x8 c) {
    return __builtin_amdgcn_wmma_f32_16x16x32_bf16(false, a, false, b, (short)0, c, false, false);
}

enum { BB = 8, CC = 512, CQ = 64, HW = 4096 };

// ---------------- kernel 0: fp32 -> bf16 convert ----------------
__global__ void cvt_f32_bf16(const float* __restrict__ s, unsigned short* __restrict__ d, int n) {
    int i = blockIdx.x * blockDim.x + threadIdx.x;
    if (i < n) d[i] = f2bf(s[i]);
}

// ---------------- kernel 1: x[B,512,4096] f32 -> xt[B,4096,512] bf16 ----------------
__global__ void transpose_x(const float* __restrict__ x, unsigned short* __restrict__ xt) {
    __shared__ unsigned short t[32][33];
    int b = blockIdx.z;
    int i0 = blockIdx.x * 32, c0 = blockIdx.y * 32;
    const float* xb = x + (size_t)b * CC * HW;
    unsigned short* xtb = xt + (size_t)b * HW * CC;
    int tx = threadIdx.x, ty = threadIdx.y; // 32 x 8
#pragma unroll
    for (int r = 0; r < 4; ++r) {
        int c = c0 + ty + r * 8;
        t[ty + r * 8][tx] = f2bf(xb[(size_t)c * HW + i0 + tx]);
    }
    __syncthreads();
#pragma unroll
    for (int r = 0; r < 4; ++r) {
        int i = i0 + ty + r * 8;
        xtb[(size_t)i * CC + c0 + tx] = t[tx][ty + r * 8];
    }
}

// ---------------- kernel 2: 5 small projections -> [B,4096,64] bf16 (transposed) ----------
// dst[m][b][i][o] = sum_c xt[b][i][c] * w_m[o][c] + bias_m[o]
__global__ void __launch_bounds__(256) proj5(
    const unsigned short* __restrict__ xt, const unsigned short* __restrict__ wb5,
    const float* __restrict__ pq_b, const float* __restrict__ pk_b,
    const float* __restrict__ cq_b, const float* __restrict__ ck_b,
    const float* __restrict__ cv_b, unsigned short* __restrict__ dstBase) {
    int lane = threadIdx.x & 31, w = threadIdx.x >> 5;
    int m = blockIdx.y, b = blockIdx.z;
    int i0 = blockIdx.x * 128 + w * 16;
    int ln15 = lane & 15;
    const float* bias = (m == 0) ? pq_b : (m == 1) ? pk_b : (m == 2) ? cq_b : (m == 3) ? ck_b : cv_b;
    const unsigned short* A = xt + ((size_t)b * HW + i0 + ln15) * CC;
    const unsigned short* Wr[4];
#pragma unroll
    for (int t = 0; t < 4; ++t)
        Wr[t] = wb5 + (size_t)m * CQ * CC + (size_t)(t * 16 + ln15) * CC;

    f32x8 acc[4];
#pragma unroll
    for (int t = 0; t < 4; ++t) acc[t] = (f32x8)0.0f;

#pragma unroll 4
    for (int kc = 0; kc < CC; kc += 32) {
        // batch all fragment loads first (one clause), then the 4 WMMAs retire with
        // descending waitcnts so WMMA t overlaps the loads for t+1..3
        bf16x16 a = loadA(A, kc, lane);
        bf16x16 bb[4];
#pragma unroll
        for (int t = 0; t < 4; ++t) bb[t] = loadB(Wr[t], kc, lane);
#pragma unroll
        for (int t = 0; t < 4; ++t) acc[t] = wmma_bf16(a, bb[t], acc[t]);
    }

    unsigned short* D = dstBase + (size_t)m * (BB * HW * CQ) + (size_t)b * HW * CQ;
    int half8 = (lane & 16) ? 8 : 0;
#pragma unroll
    for (int t = 0; t < 4; ++t) {
        int o = t * 16 + ln15;
        float bv = bias[o];
#pragma unroll
        for (int r = 0; r < 8; ++r)
            D[(size_t)(i0 + r + half8) * CQ + o] = f2bf(acc[t][r] + bv);
    }
}

// ---------------- kernel 3: v[B,512,4096] bf16 = pv_w @ x + pv_b ----------------
__global__ void __launch_bounds__(256) proj_v(
    const unsigned short* __restrict__ xt, const unsigned short* __restrict__ pvwb,
    const float* __restrict__ pv_b, unsigned short* __restrict__ vout) {
    int lane = threadIdx.x & 31, w = threadIdx.x >> 5;
    int b = blockIdx.z;
    int c0 = blockIdx.y * 128 + w * 16;
    int n0 = blockIdx.x * 64;
    int ln15 = lane & 15;
    const unsigned short* A = pvwb + (size_t)(c0 + ln15) * CC;
    const unsigned short* Xr[4];
#pragma unroll
    for (int t = 0; t < 4; ++t)
        Xr[t] = xt + (size_t)b * HW * CC + (size_t)(n0 + t * 16 + ln15) * CC;

    f32x8 acc[4];
#pragma unroll
    for (int t = 0; t < 4; ++t) acc[t] = (f32x8)0.0f;

#pragma unroll 4
    for (int kc = 0; kc < CC; kc += 32) {
        bf16x16 a = loadA(A, kc, lane);
        bf16x16 bb[4];
#pragma unroll
        for (int t = 0; t < 4; ++t) bb[t] = loadB(Xr[t], kc, lane);
#pragma unroll
        for (int t = 0; t < 4; ++t) acc[t] = wmma_bf16(a, bb[t], acc[t]);
    }

    unsigned short* V = vout + (size_t)b * CC * HW;
    int half8 = (lane & 16) ? 8 : 0;
#pragma unroll
    for (int t = 0; t < 4; ++t) {
        int j = n0 + t * 16 + ln15;
#pragma unroll
        for (int r = 0; r < 8; ++r) {
            int c = c0 + r + half8;
            V[(size_t)c * HW + j] = f2bf(acc[t][r] + pv_b[c]);
        }
    }
}

// ---------------- kernel 4: position softmax row stats l_i = m_i + log Z_i --------------
// Lane-local online softmax: each lane tracks max/sum over only the columns it owns
// (no cross-lane ops in the hot loop); the 16-lane merge happens once in the epilogue:
//   m = max_lane rm;  Z = sum_lane rs * exp(rm - m);  l = m + log Z.
__global__ void __launch_bounds__(256) pos_rowstats(
    const unsigned short* __restrict__ qT, const unsigned short* __restrict__ kT,
    float* __restrict__ lrow) {
    int lane = threadIdx.x & 31, w = threadIdx.x >> 5;
    int b = blockIdx.y;
    int i0 = blockIdx.x * 128 + w * 16;
    int ln15 = lane & 15;
    const unsigned short* qb = qT + ((size_t)b * HW + i0 + ln15) * CQ;
    bf16x16 a0 = loadA(qb, 0, lane), a1 = loadA(qb, 32, lane);
    float rm[8], rs[8];
#pragma unroll
    for (int r = 0; r < 8; ++r) { rm[r] = -1e30f; rs[r] = 0.0f; }
    const unsigned short* kb = kT + (size_t)b * HW * CQ;

#pragma unroll 2
    for (int jt = 0; jt < HW / 16; ++jt) {
        const unsigned short* kr = kb + (size_t)(jt * 16 + ln15) * CQ;
        bf16x16 b0 = loadB(kr, 0, lane);
        bf16x16 b1 = loadB(kr, 32, lane);
        f32x8 s = (f32x8)0.0f;
        s = wmma_bf16(a0, b0, s);
        s = wmma_bf16(a1, b1, s);
#pragma unroll
        for (int r = 0; r < 8; ++r) {
            float v = s[r];
            float nm = fmaxf(rm[r], v);
            rs[r] = rs[r] * __expf(rm[r] - nm) + __expf(v - nm);
            rm[r] = nm;
        }
    }
    // merge the 16 per-lane partitions of each row
#pragma unroll
    for (int r = 0; r < 8; ++r) {
        float m = rm[r];
        m = fmaxf(m, __shfl_xor(m, 1, 32));
        m = fmaxf(m, __shfl_xor(m, 2, 32));
        m = fmaxf(m, __shfl_xor(m, 4, 32));
        m = fmaxf(m, __shfl_xor(m, 8, 32));
        float z = rs[r] * __expf(rm[r] - m);
        z += __shfl_xor(z, 1, 32);
        z += __shfl_xor(z, 2, 32);
        z += __shfl_xor(z, 4, 32);
        z += __shfl_xor(z, 8, 32);
        rm[r] = m;
        rs[r] = z;
    }
    if ((lane & 15) == 0) {
        int half8 = (lane & 16) ? 8 : 0;
#pragma unroll
        for (int r = 0; r < 8; ++r)
            lrow[(size_t)b * HW + i0 + r + half8] = rm[r] + __logf(rs[r]);
    }
}

// ---------------- kernel 5: channel energy cE[b][64][64] = cq @ ck^T ----------------
__global__ void __launch_bounds__(256) chan_energy(
    const unsigned short* __restrict__ cqT, const unsigned short* __restrict__ ckT,
    float* __restrict__ cE) {
    int lane = threadIdx.x & 31, w = threadIdx.x >> 5;
    int b = blockIdx.x;
    int mt = w >> 1;
    int nt0 = (w & 1) * 2;
    int ln15 = lane & 15;
    const unsigned short* qb = cqT + (size_t)b * HW * CQ + (mt * 16 + ln15);
    const unsigned short* k0p = ckT + (size_t)b * HW * CQ + ((nt0    ) * 16 + ln15);
    const unsigned short* k1p = ckT + (size_t)b * HW * CQ + ((nt0 + 1) * 16 + ln15);
    f32x8 acc0 = (f32x8)0.0f, acc1 = (f32x8)0.0f;
    for (int kc = 0; kc < HW; kc += 32) {
        bf16x16 a  = loadA_strided(qb, CQ, kc, lane);
        bf16x16 b0 = loadB_strided(k0p, CQ, kc, lane);
        bf16x16 b1 = loadB_strided(k1p, CQ, kc, lane);
        acc0 = wmma_bf16(a, b0, acc0);
        acc1 = wmma_bf16(a, b1, acc1);
    }
    int half8 = (lane & 16) ? 8 : 0;
    float* E = cE + (size_t)b * CQ * CQ;
#pragma unroll
    for (int r = 0; r < 8; ++r) {
        int p = mt * 16 + r + half8;
        E[p * CQ + (nt0    ) * 16 + ln15] = acc0[r];
        E[p * CQ + (nt0 + 1) * 16 + ln15] = acc1[r];
    }
}

// ------------- kernel 6: softmax rows of cE, fold W2 = co_w @ c_attn (bf16) -------------
__global__ void __launch_bounds__(256) chan_attn_w2(
    const float* __restrict__ cE, const float* __restrict__ co_w,
    unsigned short* __restrict__ W2) {
    __shared__ float attn[64][64];
    int b = blockIdx.x, tid = threadIdx.x;
    if (tid < 64) {
        const float* row = cE + (size_t)b * CQ * CQ + tid * CQ;
        float m = -1e30f;
        for (int q = 0; q < 64; ++q) m = fmaxf(m, row[q]);
        float s = 0.0f;
        for (int q = 0; q < 64; ++q) { float e = __expf(row[q] - m); attn[tid][q] = e; s += e; }
        float inv = 1.0f / s;
        for (int q = 0; q < 64; ++q) attn[tid][q] *= inv;
    }
    __syncthreads();
    for (int o = tid; o < CC * CQ; o += 256) {
        int c = o >> 6, q = o & 63;
        float s = 0.0f;
#pragma unroll 8
        for (int p = 0; p < 64; ++p) s += co_w[c * CQ + p] * attn[p][q];
        W2[(size_t)b * CC * CQ + o] = f2bf(s);
    }
}

// ------------- kernel 7: out = W2 @ cv + co_b  (channel branch, output base) -------------
__global__ void __launch_bounds__(256) chan_out(
    const unsigned short* __restrict__ W2, const unsigned short* __restrict__ cvT,
    const float* __restrict__ co_b, float* __restrict__ out) {
    int lane = threadIdx.x & 31, w = threadIdx.x >> 5;
    int b = blockIdx.z;
    int c0 = blockIdx.y * 128 + w * 16;
    int n0 = blockIdx.x * 64;
    int ln15 = lane & 15;
    const unsigned short* A = W2 + (size_t)b * CC * CQ + (size_t)(c0 + ln15) * CQ;
    const unsigned short* vb = cvT + (size_t)b * HW * CQ;
    f32x8 acc[4];
#pragma unroll
    for (int t = 0; t < 4; ++t) acc[t] = (f32x8)0.0f;
#pragma unroll
    for (int kc = 0; kc < CQ; kc += 32) {
        bf16x16 a = loadA(A, kc, lane);
        bf16x16 bb[4];
#pragma unroll
        for (int t = 0; t < 4; ++t) bb[t] = loadB(vb + (size_t)(n0 + t * 16 + ln15) * CQ, kc, lane);
#pragma unroll
        for (int t = 0; t < 4; ++t) acc[t] = wmma_bf16(a, bb[t], acc[t]);
    }
    float* O = out + (size_t)b * CC * HW;
    int half8 = (lane & 16) ? 8 : 0;
#pragma unroll
    for (int t = 0; t < 4; ++t) {
        int j = n0 + t * 16 + ln15;
#pragma unroll
        for (int r = 0; r < 8; ++r) {
            int c = c0 + r + half8;
            O[(size_t)c * HW + j] = acc[t][r] + co_b[c];
        }
    }
}

// --------- kernel 8: pos_out[c,j] = sum_i v[c,i]*exp(e_ij - l_i), added into out ---------
// grid: (HW/64 j-blocks, B).  block = 256 (8 waves).  Each workgroup owns full C=512 x 64 j.
// Unrolled-by-2 K loop with FIXED LDS buffer roles (chunk ic -> buf0, ic+32 -> buf1):
// one barrier per 32-chunk, no fragment value rotation, and both chunks' v fragments are
// loaded at the top of the iteration so global loads overlap WMMAs.
__global__ void __launch_bounds__(256) pos_out_add(
    const unsigned short* __restrict__ qT, const unsigned short* __restrict__ kT,
    const unsigned short* __restrict__ vB, const float* __restrict__ lrow,
    float* __restrict__ out) {
    __shared__ unsigned short Pt[2][64 * 40];   // [buf][j 0..63][i_local 0..31], row stride 40
    int lane = threadIdx.x & 31, w = threadIdx.x >> 5;
    int b = blockIdx.y;
    int j0 = blockIdx.x * 64;
    int ih = w >> 2;        // which 16-row i half of the 32-chunk this wave generates
    int jtp = w & 3;        // which j tile this wave generates
    int cw0 = w * 64;       // this wave's 64 channel rows in the main GEMM
    int ln15 = lane & 15;
    int half8 = (lane & 16) ? 8 : 0;

    // k-tile B fragments (K = c = 64) are invariant across the whole i loop
    const unsigned short* kb = kT + ((size_t)b * HW + j0 + jtp * 16 + ln15) * CQ;
    bf16x16 kf0 = loadB(kb, 0, lane), kf1 = loadB(kb, 32, lane);

    const unsigned short* qbase = qT + (size_t)b * HW * CQ;
    const float* lb = lrow + (size_t)b * HW;

    const unsigned short* vrow[4];
#pragma unroll
    for (int ct = 0; ct < 4; ++ct)
        vrow[ct] = vB + (size_t)b * CC * HW + (size_t)(cw0 + ct * 16 + ln15) * HW;

    f32x8 acc[4][4];
#pragma unroll
    for (int ct = 0; ct < 4; ++ct)
#pragma unroll
        for (int jt = 0; jt < 4; ++jt) acc[ct][jt] = (f32x8)0.0f;

    // phase-A generator: S = q_i^T k_j (2 WMMA), P = exp(S - l_i) -> LDS[buf] transposed [j][i]
    auto genP = [&](int ic, int buf) {
        int iA = ic + ih * 16;
        const unsigned short* qr = qbase + (size_t)(iA + ln15) * CQ;
        bf16x16 a0 = loadA(qr, 0, lane), a1 = loadA(qr, 32, lane);
        f32x8 s = (f32x8)0.0f;
        s = wmma_bf16(a0, kf0, s);
        s = wmma_bf16(a1, kf1, s);
        u32x4 pk;
#pragma unroll
        for (int rr = 0; rr < 4; ++rr) {
            float e0 = __expf(s[2 * rr]     - lb[iA + 2 * rr     + half8]);
            float e1 = __expf(s[2 * rr + 1] - lb[iA + 2 * rr + 1 + half8]);
            pk[rr] = (unsigned int)f2bf(e0) | ((unsigned int)f2bf(e1) << 16);
        }
        int jrow = jtp * 16 + ln15;
        *(u32x4*)&Pt[buf][jrow * 40 + ih * 16 + half8] = pk;
    };

    genP(0, 0);   // prologue: chunk 0 into buf 0

    for (int ic = 0; ic < HW; ic += 64) {
        // both halves' v fragments: global loads issued before any barrier/WMMA
        bf16x16 Av0[4], Av1[4];
#pragma unroll
        for (int ct = 0; ct < 4; ++ct) Av0[ct] = loadA(vrow[ct], ic, lane);
#pragma unroll
        for (int ct = 0; ct < 4; ++ct) Av1[ct] = loadA(vrow[ct], ic + 32, lane);

        __syncthreads();            // buf0 (chunk ic) writes visible; prior buf1 reads done
        genP(ic + 32, 1);           // overlap: produce chunk ic+32 into buf1
        bf16x16 Bl[4];
#pragma unroll
        for (int jt = 0; jt < 4; ++jt)
            Bl[jt] = loadB(&Pt[0][(jt * 16 + ln15) * 40], 0, lane);
#pragma unroll
        for (int ct = 0; ct < 4; ++ct)
#pragma unroll
            for (int jt = 0; jt < 4; ++jt)
                acc[ct][jt] = wmma_bf16(Av0[ct], Bl[jt], acc[ct][jt]);

        __syncthreads();            // buf1 writes visible; buf0 reads done
        if (ic + 64 < HW) genP(ic + 64, 0);   // produce next iteration's buf0
#pragma unroll
        for (int jt = 0; jt < 4; ++jt)
            Bl[jt] = loadB(&Pt[1][(jt * 16 + ln15) * 40], 0, lane);
#pragma unroll
        for (int ct = 0; ct < 4; ++ct)
#pragma unroll
            for (int jt = 0; jt < 4; ++jt)
                acc[ct][jt] = wmma_bf16(Av1[ct], Bl[jt], acc[ct][jt]);
    }

    float* O = out + (size_t)b * CC * HW;
#pragma unroll
    for (int ct = 0; ct < 4; ++ct)
#pragma unroll
        for (int jt = 0; jt < 4; ++jt) {
            int j = j0 + jt * 16 + ln15;
#pragma unroll
            for (int r = 0; r < 8; ++r) {
                size_t idx = (size_t)(cw0 + ct * 16 + r + half8) * HW + j;
                O[idx] += acc[ct][jt][r];
            }
        }
}

// ---------------- host launcher ----------------
extern "C" void kernel_launch(void* const* d_in, const int* in_sizes, int n_in,
                              void* d_out, int out_size, void* d_ws, size_t ws_size,
                              hipStream_t stream) {
    const float* x    = (const float*)d_in[0];
    const float* pq_w = (const float*)d_in[1];
    const float* pq_b = (const float*)d_in[2];
    const float* pk_w = (const float*)d_in[3];
    const float* pk_b = (const float*)d_in[4];
    const float* pv_w = (const float*)d_in[5];
    const float* pv_b = (const float*)d_in[6];
    const float* cq_w = (const float*)d_in[7];
    const float* cq_b = (const float*)d_in[8];
    const float* ck_w = (const float*)d_in[9];
    const float* ck_b = (const float*)d_in[10];
    const float* cv_w = (const float*)d_in[11];
    const float* cv_b = (const float*)d_in[12];
    const float* co_w = (const float*)d_in[13];
    const float* co_b = (const float*)d_in[14];
    float* out = (float*)d_out;
    (void)in_sizes; (void)n_in; (void)out_size; (void)ws_size;

    char* p = (char*)d_ws;
    auto carve = [&](size_t bytes) { char* r = p; p += (bytes + 255) & ~(size_t)255; return r; };

    unsigned short* XT   = (unsigned short*)carve((size_t)BB * HW * CC * 2);      // x transposed bf16
    unsigned short* QT5  = (unsigned short*)carve((size_t)5 * BB * HW * CQ * 2);  // qT,kT,cqT,ckT,cvT
    unsigned short* VB   = (unsigned short*)carve((size_t)BB * CC * HW * 2);      // v bf16
    unsigned short* WB5  = (unsigned short*)carve((size_t)5 * CQ * CC * 2);       // 5 small weights bf16
    unsigned short* PVWB = (unsigned short*)carve((size_t)CC * CC * 2);           // pv_w bf16
    unsigned short* W2   = (unsigned short*)carve((size_t)BB * CC * CQ * 2);      // co_w @ c_attn bf16
    float*          LR   = (float*)carve((size_t)BB * HW * 4);                    // l_i stats
    float*          CE   = (float*)carve((size_t)BB * CQ * CQ * 4);               // channel energy

    const size_t mstride = (size_t)BB * HW * CQ;
    unsigned short* QT  = QT5 + 0 * mstride;
    unsigned short* KT  = QT5 + 1 * mstride;
    unsigned short* CQT = QT5 + 2 * mstride;
    unsigned short* CKT = QT5 + 3 * mstride;
    unsigned short* CVT = QT5 + 4 * mstride;

    // weights -> bf16
    int nw = CQ * CC;
    cvt_f32_bf16<<<(nw + 255) / 256, 256, 0, stream>>>(pq_w, WB5 + 0 * nw, nw);
    cvt_f32_bf16<<<(nw + 255) / 256, 256, 0, stream>>>(pk_w, WB5 + 1 * nw, nw);
    cvt_f32_bf16<<<(nw + 255) / 256, 256, 0, stream>>>(cq_w, WB5 + 2 * nw, nw);
    cvt_f32_bf16<<<(nw + 255) / 256, 256, 0, stream>>>(ck_w, WB5 + 3 * nw, nw);
    cvt_f32_bf16<<<(nw + 255) / 256, 256, 0, stream>>>(cv_w, WB5 + 4 * nw, nw);
    cvt_f32_bf16<<<(CC * CC + 255) / 256, 256, 0, stream>>>(pv_w, PVWB, CC * CC);

    // x -> xt bf16
    transpose_x<<<dim3(HW / 32, CC / 32, BB), dim3(32, 8), 0, stream>>>(x, XT);

    // projections
    proj5<<<dim3(HW / 128, 5, BB), 256, 0, stream>>>(XT, WB5, pq_b, pk_b, cq_b, ck_b, cv_b, QT5);
    proj_v<<<dim3(HW / 64, CC / 128, BB), 256, 0, stream>>>(XT, PVWB, pv_b, VB);

    // position softmax row stats (flash pass 1)
    pos_rowstats<<<dim3(HW / 128, BB), 256, 0, stream>>>(QT, KT, LR);

    // channel branch (tiny): energy -> softmax+fold -> output base
    chan_energy<<<dim3(BB), 256, 0, stream>>>(CQT, CKT, CE);
    chan_attn_w2<<<dim3(BB), 256, 0, stream>>>(CE, co_w, W2);
    chan_out<<<dim3(HW / 64, CC / 128, BB), 256, 0, stream>>>(W2, CVT, co_b, out);

    // position branch (flash pass 2): regenerate P tiles in LDS, big v @ P GEMM, += into out
    pos_out_add<<<dim3(HW / 64, BB), 256, 0, stream>>>(QT, KT, VB, LR, out);
}
